// PCAM_Module_13408887898893
// MI455X (gfx1250) — matrix operand order repaired and checked by reading, not compiled, not run on hardware
//
#include <hip/hip_runtime.h>
#include <stdint.h>

#define B_ 4
#define C_ 256
#define N_ 4096   // 64*64
#define D_ 32

typedef __attribute__((ext_vector_type(16))) __bf16 v16bf;
typedef __attribute__((ext_vector_type(8)))  float  v8f;

union U4 { v16bf v; uint4 q[2]; unsigned short h[16]; };

static __device__ __forceinline__ unsigned short f2bf(float f) {
  union { float f; uint32_t u; } a; a.f = f;
  uint32_t r = a.u + 0x7FFFu + ((a.u >> 16) & 1u);
  return (unsigned short)(r >> 16);
}

#define WMMA_BF16(A, Bm, Cc) \
  __builtin_amdgcn_wmma_f32_16x16x32_bf16(false, (A), false, (Bm), (short)0, (Cc), false, false)

// ---- CDNA5 async global->LDS path (ASYNCcnt), with synchronous fallback ----
#if defined(__gfx1250__) && __has_builtin(__builtin_amdgcn_global_load_async_to_lds_b128)
#define HAVE_ASYNC_LDS 1
typedef int vint4_t __attribute__((ext_vector_type(4)));
typedef __attribute__((address_space(3))) vint4_t lds_vint4_t;
static __device__ __forceinline__ void async_cp16(const unsigned short* gsrc, unsigned short* ldst) {
  __builtin_amdgcn_global_load_async_to_lds_b128((vint4_t*)gsrc, (lds_vint4_t*)ldst, 0, 0);
}
static __device__ __forceinline__ void wait_async0() {
#if __has_builtin(__builtin_amdgcn_s_wait_asynccnt)
  __builtin_amdgcn_s_wait_asynccnt(0);
#else
  asm volatile("s_wait_asynccnt 0x0" ::: "memory");
#endif
}
#else
#define HAVE_ASYNC_LDS 0
static __device__ __forceinline__ void async_cp16(const unsigned short* gsrc, unsigned short* ldst) {
  *(uint4*)ldst = *(const uint4*)gsrc;
}
static __device__ __forceinline__ void wait_async0() {}
#endif

// ---------------------------------------------------------------- weights->bf16
__global__ void cvt_weights(const float* __restrict__ wq, const float* __restrict__ wk,
                            const float* __restrict__ wv,
                            unsigned short* __restrict__ wqb, unsigned short* __restrict__ wkb,
                            unsigned short* __restrict__ wvb) {
  int i = blockIdx.x * 256 + threadIdx.x;
  if (i < D_ * C_) { wqb[i] = f2bf(wq[i]); wkb[i] = f2bf(wk[i]); }
  if (i < C_ * C_) { wvb[i] = f2bf(wv[i]); }
}

// ---------------------------------------------------------------- Q / K^T projection
// Q[b][n][d] = sum_c x[b][c][n] * wq[d][c] + bq[d]      (stored [n][d] bf16)
// KT[b][n][d] = sum_c y[b][c][n] * wk[d][c] + bk[d]     (stored [n][d] bf16)
__global__ __launch_bounds__(128, 1)
void proj_qk(const float* __restrict__ x, const float* __restrict__ y,
             const unsigned short* __restrict__ wqb, const unsigned short* __restrict__ wkb,
             const float* __restrict__ bq, const float* __restrict__ bk,
             unsigned short* __restrict__ Qb, unsigned short* __restrict__ Kb) {
  __shared__ __align__(16) unsigned short Xlds[64 * 40];  // x^T tile [n][c], padded rows
  __shared__ __align__(16) unsigned short Ylds[64 * 40];
  const int tid = threadIdx.x;
  const int b = blockIdx.y;
  const int n0 = blockIdx.x * 64;
  const int lane = tid & 31, wvid = tid >> 5;
  const int col = lane & 15, half = lane >> 4;
  const int cb = half * 8;

  v8f accQ[2], accK[2];
#pragma unroll
  for (int t = 0; t < 2; ++t) { accQ[t] = (v8f){0.f,0.f,0.f,0.f,0.f,0.f,0.f,0.f}; accK[t] = accQ[t]; }

  for (int kk = 0; kk < C_; kk += 32) {
#pragma unroll
    for (int e = 0; e < 16; ++e) {
      int idx = e * 128 + tid;        // 0..2047
      int ci = idx >> 6, nn = idx & 63;
      size_t g = ((size_t)b * C_ + kk + ci) * N_ + n0 + nn;
      Xlds[nn * 40 + ci] = f2bf(x[g]);
      Ylds[nn * 40 + ci] = f2bf(y[g]);
    }
    __syncthreads();

    const int arow = (wvid * 16 + col) * 40;
    U4 ax, ay;
    ax.q[0] = *(const uint4*)&Xlds[arow + cb];
    ax.q[1] = *(const uint4*)&Xlds[arow + cb + 16];
    ay.q[0] = *(const uint4*)&Ylds[arow + cb];
    ay.q[1] = *(const uint4*)&Ylds[arow + cb + 16];

    const int cstart = kk + half * 8;
#pragma unroll
    for (int dt = 0; dt < 2; ++dt) {
      U4 bwq, bwk;
      const unsigned short* wr = wqb + (size_t)(dt * 16 + col) * C_ + cstart;
      bwq.q[0] = *(const uint4*)wr; bwq.q[1] = *(const uint4*)(wr + 16);
      const unsigned short* kr = wkb + (size_t)(dt * 16 + col) * C_ + cstart;
      bwk.q[0] = *(const uint4*)kr; bwk.q[1] = *(const uint4*)(kr + 16);
      accQ[dt] = WMMA_BF16(ax.v, bwq.v, accQ[dt]);
      accK[dt] = WMMA_BF16(ay.v, bwk.v, accK[dt]);
    }
    __syncthreads();
  }
#pragma unroll
  for (int dt = 0; dt < 2; ++dt) {
    const int d = dt * 16 + col;
    const float bqv = bq[d], bkv = bk[d];
#pragma unroll
    for (int r = 0; r < 8; ++r) {
      int n = n0 + wvid * 16 + r + half * 8;
      size_t o = ((size_t)b * N_ + n) * D_ + d;
      Qb[o] = f2bf(accQ[dt][r] + bqv);
      Kb[o] = f2bf(accK[dt][r] + bkv);
    }
  }
}

// ---------------------------------------------------------------- V projection
// V[b][c][n] = sum_ci wv[c][ci] * y[b][ci][n] + bv[c]   (stored [c][n] bf16)
__global__ __launch_bounds__(128, 1)
void proj_v(const float* __restrict__ y, const unsigned short* __restrict__ wvb,
            const float* __restrict__ bv, unsigned short* __restrict__ Vb) {
  __shared__ __align__(16) unsigned short Ylds[64 * 40];  // y^T tile [n][ci]
  const int tid = threadIdx.x;
  const int b = blockIdx.z;
  const int c0 = blockIdx.y * 64;
  const int n0 = blockIdx.x * 64;
  const int lane = tid & 31, wvid = tid >> 5;
  const int col = lane & 15, half = lane >> 4;
  const int cb = half * 8;

  v8f accV[4];
#pragma unroll
  for (int t = 0; t < 4; ++t) accV[t] = (v8f){0.f,0.f,0.f,0.f,0.f,0.f,0.f,0.f};

  for (int kk = 0; kk < C_; kk += 32) {
#pragma unroll
    for (int e = 0; e < 16; ++e) {
      int idx = e * 128 + tid;
      int ci = idx >> 6, nn = idx & 63;
      Ylds[nn * 40 + ci] = f2bf(y[((size_t)b * C_ + kk + ci) * N_ + n0 + nn]);
    }
    __syncthreads();

    const unsigned short* ar = wvb + (size_t)(c0 + wvid * 16 + col) * C_ + kk + cb;
    U4 aw;
    aw.q[0] = *(const uint4*)ar; aw.q[1] = *(const uint4*)(ar + 16);
#pragma unroll
    for (int t = 0; t < 4; ++t) {
      U4 by;
      const unsigned short* yr = &Ylds[(t * 16 + col) * 40 + cb];
      by.q[0] = *(const uint4*)yr; by.q[1] = *(const uint4*)(yr + 16);
      accV[t] = WMMA_BF16(aw.v, by.v, accV[t]);
    }
    __syncthreads();
  }
  float bvv[8];
#pragma unroll
  for (int r = 0; r < 8; ++r) bvv[r] = bv[c0 + wvid * 16 + r + half * 8];
#pragma unroll
  for (int t = 0; t < 4; ++t)
#pragma unroll
    for (int r = 0; r < 8; ++r) {
      int c = c0 + wvid * 16 + r + half * 8;
      int n = n0 + t * 16 + col;
      Vb[((size_t)b * C_ + c) * N_ + n] = f2bf(accV[t][r] + bvv[r]);
    }
}

// ---------------------------------------------------------------- fused flash attention
// Per wave: one 16-row q tile, all 256 channels. S^T = K.Q^T, online softmax,
// O^T = V^T.P^T. WG = 4 waves; K/V m-slabs double-buffered in LDS and filled
// with GLOBAL_LOAD_ASYNC_TO_LDS (ASYNCcnt) when available. grid (N/64, B).
__global__ __launch_bounds__(128, 1)
void attn_flash(const float* __restrict__ x, const float* __restrict__ gamma,
                const unsigned short* __restrict__ Qb, const unsigned short* __restrict__ Kb,
                const unsigned short* __restrict__ Vb, float* __restrict__ out) {
  __shared__ __align__(16) unsigned short Klds[2][32 * 40];
  __shared__ __align__(16) unsigned short Vlds[2][256 * 40];
  const int tid = threadIdx.x;
  const int b = blockIdx.y;
  const int lane = tid & 31, wvid = tid >> 5;
  const int q0 = blockIdx.x * 64 + wvid * 16;
  const int col = lane & 15, half = lane >> 4;
  const int cb = half * 8;

  // preload Q B-operand: lane = col q, rows d = half*8..+7 and 16+half*8..+7 (in Qb[n][d])
  U4 qB;
  {
    const unsigned short* qr = Qb + ((size_t)b * N_ + q0 + col) * D_ + cb;
    qB.q[0] = *(const uint4*)qr; qB.q[1] = *(const uint4*)(qr + 16);
  }

  v8f acc[16];
#pragma unroll
  for (int t = 0; t < 16; ++t) acc[t] = (v8f){0.f,0.f,0.f,0.f,0.f,0.f,0.f,0.f};
  const v8f zf = (v8f){0.f,0.f,0.f,0.f,0.f,0.f,0.f,0.f};
  float mrun = -3.0e38f, lrun = 0.f;

  const int krow = tid >> 2, kpart = tid & 3;
  auto issue_slab = [&](int m0, int bufi) {
    // K slab: 32 rows x 32 d bf16 = 128 quads (one per thread)
    async_cp16(Kb + ((size_t)b * N_ + m0 + krow) * D_ + kpart * 8,
               &Klds[bufi][krow * 40 + kpart * 8]);
    // V slab: 256 rows x 32 m bf16 = 1024 quads (8 per thread)
#pragma unroll
    for (int p = 0; p < 8; ++p) {
      int id = p * 128 + tid;
      int row = id >> 2, part = id & 3;
      async_cp16(Vb + ((size_t)b * C_ + row) * N_ + m0 + part * 8,
                 &Vlds[bufi][row * 40 + part * 8]);
    }
  };

  issue_slab(0, 0);

#pragma unroll 1
  for (int it = 0; it < N_ / 32; ++it) {
    const int bufi = it & 1;
    wait_async0();        // own async copies of slab `it` landed in LDS
    __syncthreads();      // all waves' copies landed; all reads of other buffer done
    if (it + 1 < N_ / 32) issue_slab((it + 1) * 32, bufi ^ 1);

    const unsigned short* Kp = Klds[bufi];
    const unsigned short* Vp = Vlds[bufi];

    // ---- S^T tiles: rows m, cols q
    U4 ka0, ka1;
    ka0.q[0] = *(const uint4*)&Kp[col * 40 + cb];
    ka0.q[1] = *(const uint4*)&Kp[col * 40 + cb + 16];
    ka1.q[0] = *(const uint4*)&Kp[(16 + col) * 40 + cb];
    ka1.q[1] = *(const uint4*)&Kp[(16 + col) * 40 + cb + 16];
    v8f st0 = WMMA_BF16(ka0.v, qB.v, zf);
    v8f st1 = WMMA_BF16(ka1.v, qB.v, zf);

    // ---- online softmax (row = q = lane%16; lane^16 holds the other m rows)
    float mx = st0[0];
#pragma unroll
    for (int r = 0; r < 8; ++r) { mx = fmaxf(mx, st0[r]); mx = fmaxf(mx, st1[r]); }
    mx = fmaxf(mx, __shfl_xor(mx, 16));
    float mnew = fmaxf(mrun, mx);
    float alpha = __expf(mrun - mnew);
    float p0[8], p1[8], ssum = 0.f;
#pragma unroll
    for (int r = 0; r < 8; ++r) {
      p0[r] = __expf(st0[r] - mnew);
      p1[r] = __expf(st1[r] - mnew);
      ssum += p0[r] + p1[r];
    }
    ssum += __shfl_xor(ssum, 16);
    lrun = lrun * alpha + ssum;
    mrun = mnew;
#pragma unroll
    for (int t = 0; t < 16; ++t)
#pragma unroll
      for (int r = 0; r < 8; ++r) acc[t][r] *= alpha;

    // ---- build P^T B-operand (rows m 0..31, col q = lane%16)
    U4 pb;
#pragma unroll
    for (int r = 0; r < 8; ++r) {
      float ex = __shfl_xor(half ? p0[r] : p1[r], 16);
      float lo = half ? ex : p0[r];
      float hi = half ? p1[r] : ex;
      pb.h[r] = f2bf(lo);
      pb.h[8 + r] = f2bf(hi);
    }

    // ---- O^T += V^T tile x P^T, all 16 c-tiles
#pragma unroll
    for (int t = 0; t < 16; ++t) {
      U4 va;
      const unsigned short* vr = &Vp[(t * 16 + col) * 40 + half * 16];
      va.q[0] = *(const uint4*)vr; va.q[1] = *(const uint4*)(vr + 8);
      acc[t] = WMMA_BF16(va.v, pb.v, acc[t]);
    }
  }

  // ---- epilogue: out = gamma * O/l + x
  const float inv = 1.0f / lrun;
  const float g = gamma[0];
  const int q = q0 + col;
#pragma unroll
  for (int t = 0; t < 16; ++t)
#pragma unroll
    for (int r = 0; r < 8; ++r) {
      int c = t * 16 + r + half * 8;
      size_t idx = ((size_t)b * C_ + c) * N_ + q;
      out[idx] = g * (acc[t][r] * inv) + x[idx];
    }
}

// ---------------------------------------------------------------- launcher
extern "C" void kernel_launch(void* const* d_in, const int* in_sizes, int n_in,
                              void* d_out, int out_size, void* d_ws, size_t ws_size,
                              hipStream_t stream) {
  const float* x     = (const float*)d_in[0];
  const float* y     = (const float*)d_in[1];
  const float* wq    = (const float*)d_in[2];
  const float* bq    = (const float*)d_in[3];
  const float* wk    = (const float*)d_in[4];
  const float* bk    = (const float*)d_in[5];
  const float* wv    = (const float*)d_in[6];
  const float* bv    = (const float*)d_in[7];
  const float* gamma = (const float*)d_in[8];
  float* out = (float*)d_out;

  unsigned short* ws  = (unsigned short*)d_ws;
  unsigned short* wqb = ws;
  unsigned short* wkb = wqb + D_ * C_;
  unsigned short* wvb = wkb + D_ * C_;
  unsigned short* Qb  = wvb + C_ * C_;
  unsigned short* Kb  = Qb + (size_t)B_ * N_ * D_;
  unsigned short* Vb  = Kb + (size_t)B_ * N_ * D_;

  cvt_weights<<<dim3((C_ * C_) / 256), dim3(256), 0, stream>>>(wq, wk, wv, wqb, wkb, wvb);
  proj_qk<<<dim3(N_ / 64, B_), dim3(128), 0, stream>>>(x, y, wqb, wkb, bq, bk, Qb, Kb);
  proj_v<<<dim3(N_ / 64, C_ / 64, B_), dim3(128), 0, stream>>>(y, wvb, bv, Vb);
  attn_flash<<<dim3(N_ / 64, B_), dim3(128), 0, stream>>>(x, gamma, Qb, Kb, Vb, out);
}
